// DrumPerfLSTMDecoder_35751307772067
// MI455X (gfx1250) — matrix operand order repaired and checked by reading, not compiled, not run on hardware
//
#include <hip/hip_runtime.h>
#include <hip/hip_bf16.h>
#include <math.h>

// ---- problem dims ----
#define LL 281
#define BB 256
#define VV 512
#define HH 1024
#define ZZ 256
#define G4H (4 * HH)   // 4096
#define KCAT (2 * HH)  // 2048

typedef __attribute__((ext_vector_type(16))) __bf16 v16bf;
typedef __attribute__((ext_vector_type(8)))  __bf16 v8bf;
typedef __attribute__((ext_vector_type(8)))  float  v8f;

// ---------------------------------------------------------------------------
// WMMA fragment loaders (CDNA5 16-bit A 16x32 / B 32x16 lane layouts)
// A: row-major [*,1024]; lane m=lane&15 holds row mBase+m,
//    lane<16 -> K 0..7 and 16..23 ; lane>=16 -> K 8..15 and 24..31
// B: W row-major [N,K]; lane holds column nBase+(lane&15),
//    lane<16 -> K 0..7 and 16..23 ; lane>=16 -> K 8..15 and 24..31
// ---------------------------------------------------------------------------
__device__ __forceinline__ v16bf load_a_frag(const __bf16* A, int ldA,
                                             int mBase, int kBase, int lane) {
    int row = mBase + (lane & 15);
    int kh  = (lane >> 4) * 8;
    const __bf16* p = A + (size_t)row * ldA + kBase + kh;
    v8bf lo = *reinterpret_cast<const v8bf*>(p);
    v8bf hi = *reinterpret_cast<const v8bf*>(p + 16);
    return __builtin_shufflevector(lo, hi, 0,1,2,3,4,5,6,7,8,9,10,11,12,13,14,15);
}

__device__ __forceinline__ v16bf load_b_frag(const __bf16* W, int ldW,
                                             int nBase, int kBase, int lane) {
    int col = nBase + (lane & 15);
    int ko  = (lane >> 4) * 8;
    const __bf16* p = W + (size_t)col * ldW + kBase + ko;
    v8bf lo = *reinterpret_cast<const v8bf*>(p);
    v8bf hi = *reinterpret_cast<const v8bf*>(p + 16);
    return __builtin_shufflevector(lo, hi, 0,1,2,3,4,5,6,7,8,9,10,11,12,13,14,15);
}

// ---------------------------------------------------------------------------
// GEMM: out[B,N] = concat(A0,A1)[B,K] * W[N,K]^T + biasA + biasB
// A0 covers k in [0,1024), A1 covers k in [1024,K). ldA fixed at 1024.
// Block = 256 threads = 8 waves; wave computes a 32(M) x 64(N) tile:
//   2 A-frags x 4 B-frags -> 8 wmmas per 32-wide K chunk (1.5 loads/wmma).
// grid = (N/512, B/32)
// ---------------------------------------------------------------------------
__global__ void __launch_bounds__(256)
gemm_bf16_wmma(const __bf16* __restrict__ A0, const __bf16* __restrict__ A1,
               const __bf16* __restrict__ W, int K, int N,
               const float* __restrict__ biasA, const float* __restrict__ biasB,
               float* __restrict__ out) {
    const int lane  = threadIdx.x & 31;
    const int wave  = threadIdx.x >> 5;
    const int mBase = blockIdx.y * 32;
    const int nBase = blockIdx.x * 512 + wave * 64;

    v8f zero = {0.f, 0.f, 0.f, 0.f, 0.f, 0.f, 0.f, 0.f};
    v8f acc00 = zero, acc01 = zero, acc02 = zero, acc03 = zero;  // rows mBase..mBase+15
    v8f acc10 = zero, acc11 = zero, acc12 = zero, acc13 = zero;  // rows mBase+16..mBase+31

    for (int kb = 0; kb < K; kb += 32) {
        const __bf16* Asrc = (kb < HH) ? A0 : A1;
        int kloc = (kb < HH) ? kb : (kb - HH);
        v16bf a0 = load_a_frag(Asrc, HH, mBase +  0, kloc, lane);
        v16bf a1 = load_a_frag(Asrc, HH, mBase + 16, kloc, lane);
        v16bf b0 = load_b_frag(W, K, nBase +  0, kb, lane);
        v16bf b1 = load_b_frag(W, K, nBase + 16, kb, lane);
        v16bf b2 = load_b_frag(W, K, nBase + 32, kb, lane);
        v16bf b3 = load_b_frag(W, K, nBase + 48, kb, lane);
        acc00 = __builtin_amdgcn_wmma_f32_16x16x32_bf16(false, a0, false, b0, (short)0, acc00, false, false);
        acc10 = __builtin_amdgcn_wmma_f32_16x16x32_bf16(false, a1, false, b0, (short)0, acc10, false, false);
        acc01 = __builtin_amdgcn_wmma_f32_16x16x32_bf16(false, a0, false, b1, (short)0, acc01, false, false);
        acc11 = __builtin_amdgcn_wmma_f32_16x16x32_bf16(false, a1, false, b1, (short)0, acc11, false, false);
        acc02 = __builtin_amdgcn_wmma_f32_16x16x32_bf16(false, a0, false, b2, (short)0, acc02, false, false);
        acc12 = __builtin_amdgcn_wmma_f32_16x16x32_bf16(false, a1, false, b2, (short)0, acc12, false, false);
        acc03 = __builtin_amdgcn_wmma_f32_16x16x32_bf16(false, a0, false, b3, (short)0, acc03, false, false);
        acc13 = __builtin_amdgcn_wmma_f32_16x16x32_bf16(false, a1, false, b3, (short)0, acc13, false, false);
    }

    // C/D layout: VGPR r -> M = tileM + r + 8*(lane>>4), N = nTile + (lane&15)
    const int nCol  = lane & 15;
    const int mHalf = (lane >> 4) * 8;
    v8f accs[2][4] = {{acc00, acc01, acc02, acc03}, {acc10, acc11, acc12, acc13}};
#pragma unroll
    for (int mi = 0; mi < 2; ++mi) {
#pragma unroll
        for (int s = 0; s < 4; ++s) {
            int n = nBase + s * 16 + nCol;
            float bias = (biasA ? biasA[n] : 0.f) + (biasB ? biasB[n] : 0.f);
#pragma unroll
            for (int r = 0; r < 8; ++r) {
                int m = mBase + mi * 16 + mHalf + r;
                out[(size_t)m * N + n] = accs[mi][s][r] + bias;
            }
        }
    }
}

// ---------------------------------------------------------------------------
// Helper kernels
// ---------------------------------------------------------------------------
__global__ void pack_wcat(const float* __restrict__ Wi, const float* __restrict__ Wh,
                          __bf16* __restrict__ dst, int total) {  // dst [4096,2048]
    for (int i = blockIdx.x * blockDim.x + threadIdx.x; i < total;
         i += gridDim.x * blockDim.x) {
        int n = i / KCAT, k = i % KCAT;
        float v = (k < HH) ? Wi[(size_t)n * HH + k] : Wh[(size_t)n * HH + (k - HH)];
        dst[i] = (__bf16)v;
    }
}

__global__ void f32_to_bf16(const float* __restrict__ src, __bf16* __restrict__ dst, int total) {
    for (int i = blockIdx.x * blockDim.x + threadIdx.x; i < total;
         i += gridDim.x * blockDim.x)
        dst[i] = (__bf16)src[i];
}

__global__ void find_hot(const float* __restrict__ targets, int* __restrict__ idx, int total) {
    for (int i = blockIdx.x * blockDim.x + threadIdx.x; i < total;
         i += gridDim.x * blockDim.x) {
        const float* r = targets + (size_t)i * VV;
        int best = 0;
        for (int j = 0; j < VV; ++j)
            if (r[j] > 0.5f) { best = j; break; }
        idx[i] = best;
    }
}

__global__ void init_h(const float* __restrict__ z, const float* __restrict__ Whz,
                       const float* __restrict__ bhz,
                       __bf16* __restrict__ h, float* __restrict__ c) {
    int i = blockIdx.x * blockDim.x + threadIdx.x;  // B*H threads
    if (i >= BB * HH) return;
    int b = i / HH, j = i % HH;
    const float* zr = z + (size_t)b * ZZ;
    const float* wr = Whz + (size_t)j * ZZ;
    float s = bhz[j];
    for (int k = 0; k < ZZ; ++k) s += zr[k] * wr[k];
    h[i] = (__bf16)tanhf(s);
    c[i] = 0.f;
}

__global__ void xproj_kernel(const float* __restrict__ Wp, const float* __restrict__ bp,
                             const int* __restrict__ idxRow, int useIdx,
                             __bf16* __restrict__ xp) {
    int i = blockIdx.x * blockDim.x + threadIdx.x;  // B*H
    if (i >= BB * HH) return;
    int b = i / HH, h = i % HH;
    float v = bp[h];
    if (useIdx) v += Wp[(size_t)h * VV + idxRow[b]];
    xp[i] = (__bf16)tanhf(v);
}

__device__ __forceinline__ float sigmoidf(float x) { return 1.f / (1.f + expf(-x)); }

__global__ void lstm_cell(const float* __restrict__ g, float* __restrict__ c,
                          __bf16* __restrict__ h) {
    int i = blockIdx.x * blockDim.x + threadIdx.x;  // B*H
    if (i >= BB * HH) return;
    int b = i / HH, j = i % HH;
    const float* gr = g + (size_t)b * G4H;
    float ig = sigmoidf(gr[j]);
    float fg = sigmoidf(gr[HH + j]);
    float gg = tanhf(gr[2 * HH + j]);
    float og = sigmoidf(gr[3 * HH + j]);
    float cn = fg * c[i] + ig * gg;
    c[i] = cn;
    h[i] = (__bf16)(og * tanhf(cn));
}

__global__ void argmax_kernel(const float* __restrict__ logits, float* __restrict__ samples) {
    int b = blockIdx.x * blockDim.x + threadIdx.x;
    if (b >= BB) return;
    const float* r = logits + (size_t)b * VV;
    int best = 0; float bv = r[0];
    for (int j = 1; j < VV; ++j) {
        float v = r[j];
        if (v > bv) { bv = v; best = j; }
    }
    samples[b] = (float)best;
}

// ---------------------------------------------------------------------------
extern "C" void kernel_launch(void* const* d_in, const int* in_sizes, int n_in,
                              void* d_out, int out_size, void* d_ws, size_t ws_size,
                              hipStream_t stream) {
    const float* z    = (const float*)d_in[0];
    const float* tgt  = (const float*)d_in[1];
    const float* Wp   = (const float*)d_in[2];
    const float* bp   = (const float*)d_in[3];
    const float* Wi0  = (const float*)d_in[4];
    const float* Wh0  = (const float*)d_in[5];
    const float* bi0  = (const float*)d_in[6];
    const float* bh0  = (const float*)d_in[7];
    const float* Wi1  = (const float*)d_in[8];
    const float* Wh1  = (const float*)d_in[9];
    const float* bi1  = (const float*)d_in[10];
    const float* bh1  = (const float*)d_in[11];
    const float* Whz0 = (const float*)d_in[12];
    const float* bhz0 = (const float*)d_in[13];
    const float* Whz1 = (const float*)d_in[14];
    const float* bhz1 = (const float*)d_in[15];
    const float* WoutF = (const float*)d_in[16];
    const float* bout  = (const float*)d_in[17];

    // workspace layout (256B aligned slabs)
    char* ws = (char*)d_ws;
    size_t off = 0;
    auto alloc = [&](size_t bytes) { void* p = ws + off; off = (off + bytes + 255) & ~(size_t)255; return p; };
    __bf16* W0cat = (__bf16*)alloc((size_t)G4H * KCAT * 2);  // 16.8 MB
    __bf16* W1cat = (__bf16*)alloc((size_t)G4H * KCAT * 2);  // 16.8 MB
    __bf16* WoutB = (__bf16*)alloc((size_t)VV * HH * 2);     //  1.0 MB
    int*    idx   = (int*)   alloc((size_t)LL * BB * 4);
    __bf16* xp    = (__bf16*)alloc((size_t)BB * HH * 2);
    __bf16* h0    = (__bf16*)alloc((size_t)BB * HH * 2);
    __bf16* h1    = (__bf16*)alloc((size_t)BB * HH * 2);
    float*  c0    = (float*) alloc((size_t)BB * HH * 4);
    float*  c1    = (float*) alloc((size_t)BB * HH * 4);
    float*  gbuf  = (float*) alloc((size_t)BB * G4H * 4);    //  4.2 MB

    float* samples = (float*)d_out;                          // [L,B]
    float* logits  = (float*)d_out + (size_t)LL * BB;        // [L,B,V]

    const int EBLK = 256;
    const int nBH  = BB * HH;

    // one-time prep (re-run every call; deterministic)
    pack_wcat<<<2048, EBLK, 0, stream>>>(Wi0, Wh0, W0cat, G4H * KCAT);
    pack_wcat<<<2048, EBLK, 0, stream>>>(Wi1, Wh1, W1cat, G4H * KCAT);
    f32_to_bf16<<<1024, EBLK, 0, stream>>>(WoutF, WoutB, VV * HH);
    find_hot<<<(LL * BB + EBLK - 1) / EBLK, EBLK, 0, stream>>>(tgt, idx, LL * BB);
    init_h<<<nBH / EBLK, EBLK, 0, stream>>>(z, Whz0, bhz0, h0, c0);
    init_h<<<nBH / EBLK, EBLK, 0, stream>>>(z, Whz1, bhz1, h1, c1);

    dim3 gGates(G4H / 512, BB / 32);  // (8,8)
    dim3 gLogit(VV / 512, BB / 32);   // (1,8)

    for (int t = 0; t < LL; ++t) {
        // input projection (one-hot gather), step 0 input is zeros
        const int* ir = (t > 0) ? (idx + (size_t)(t - 1) * BB) : idx;
        xproj_kernel<<<nBH / EBLK, EBLK, 0, stream>>>(Wp, bp, ir, t > 0 ? 1 : 0, xp);

        // layer 0: g = [xp|h0] * [Wi0|Wh0]^T + bi0 + bh0
        gemm_bf16_wmma<<<gGates, 256, 0, stream>>>(xp, h0, W0cat, KCAT, G4H, bi0, bh0, gbuf);
        lstm_cell<<<nBH / EBLK, EBLK, 0, stream>>>(gbuf, c0, h0);

        // layer 1: g = [h0|h1] * [Wi1|Wh1]^T + bi1 + bh1
        gemm_bf16_wmma<<<gGates, 256, 0, stream>>>(h0, h1, W1cat, KCAT, G4H, bi1, bh1, gbuf);
        lstm_cell<<<nBH / EBLK, EBLK, 0, stream>>>(gbuf, c1, h1);

        // logits = h1 * Wout^T + bout  (K = 1024, single A source)
        float* lg = logits + (size_t)t * BB * VV;
        gemm_bf16_wmma<<<gLogit, 256, 0, stream>>>(h1, h1, WoutB, HH, VV, bout, nullptr, lg);
        argmax_kernel<<<1, BB, 0, stream>>>(lg, samples + (size_t)t * BB);
    }
}
